// Realnvp_26482768347693
// MI455X (gfx1250) — matrix operand order repaired and checked by reading, not compiled, not run on hardware
//
#include <hip/hip_runtime.h>
#include <hip/hip_bf16.h>
#include <math.h>

typedef __attribute__((ext_vector_type(16))) _Float16     v16h;
typedef __attribute__((ext_vector_type(8)))  _Float16     v8h;
typedef __attribute__((ext_vector_type(8)))  float        v8f;
typedef __attribute__((ext_vector_type(8)))  unsigned int u32x8;

#define NBATCH 65536
#define DDIM   512
#define NHALF  256
#define NHID   256
#define NLAYER 8

// ---------------------------------------------------------------------------
// Weight packing: fp32 (256x256 row-major, [k][n]) -> f16 in WMMA B-operand
// per-lane register layout.  B operand (32x16): lane (n + 16*half) holds
// dword p = halves k = 32*kb + 8*half + 2*p + 8*(p>=4) (+lo/hi).  Each (nb,kb) tile is
// 32 lanes x 32 bytes contiguous => a wave fetches a whole B tile as 1 KB of
// coalesced global_load_b128s straight from L2 (weights stay L2-resident).
// Matrix order in d_ws: mat = i*4 + {0:Ws1, 1:Ws2, 2:Wt1, 3:Wt2}.
// ---------------------------------------------------------------------------
__global__ __launch_bounds__(256) void pack_weights_kernel(
    const float* __restrict__ Ws1, const float* __restrict__ Ws2,
    const float* __restrict__ Wt1, const float* __restrict__ Wt2,
    _Float16* __restrict__ wpk)
{
    int gid  = blockIdx.x * 256 + threadIdx.x;  // 0 .. 524287
    int flat = gid * 4;
    int mat  = flat >> 16;                      // 0..31
    int e0   = flat & 65535;
    int i = mat >> 2, t = mat & 3;
    const float* src = (t == 0 ? Ws1 : t == 1 ? Ws2 : t == 2 ? Wt1 : Wt2)
                       + (size_t)i * 65536 + e0;
    float4 w = *(const float4*)src;
    float wv[4] = {w.x, w.y, w.z, w.w};
#pragma unroll
    for (int j = 0; j < 4; ++j) {
        int e = e0 + j;
        int k = e >> 8, n = e & 255;
        int nb = n >> 4, nn = n & 15;
        int kb = k >> 5, kl = k & 31;
        int half = (kl >> 3) & 1;
        int p    = ((kl & 7) >> 1) + ((kl >> 4) << 2);
        int lohi = kl & 1;
        int lane2 = nn + 16 * half;
        size_t dst = ((size_t)mat << 16)
                   + ((size_t)(((nb * 8 + kb) * 32 + lane2) * 8 + p) << 1) + lohi;
        wpk[dst] = (_Float16)wv[j];
    }
}

// A-operand (16x32 f16) K offset for element pair p, lane half (ISA 7.12.2)
__device__ __forceinline__ int ka_base(int p, int half) {
    return ((p < 4) ? 2 * p : 2 * p + 8) + 8 * half;  // always even
}

// Pre-packed B tile: 32 contiguous bytes per lane at a constant byte offset
// from a per-matrix lane pointer -> 2x global_load_b128 with immediate offset.
__device__ __forceinline__ v16h load_bw(const char* base_lane, int tile_bytes) {
    u32x8 v = *(const u32x8*)(base_lane + tile_bytes);
    return __builtin_bit_cast(v16h, v);
}

// A tile from LDS (per-lane contiguous) -> 2x ds_load_b128.
__device__ __forceinline__ v16h load_lds_a(const u32x8* s, int idx) {
    return __builtin_bit_cast(v16h, s[idx]);
}

__device__ __forceinline__ float tanh_fast(float x) {
    x = fminf(fmaxf(x, -15.0f), 15.0f);
    float e2 = __expf(2.0f * x);
    return (e2 - 1.0f) / (e2 + 1.0f);
}

// ---------------------------------------------------------------------------
// Main fused kernel.  One wave pair (64 threads) owns 16 batch rows through
// all 8 coupling layers.  E (conditioner, even z columns) is parked in LDS in
// A-operand layout (producer lane == consumer lane, so no barrier needed);
// only O (odd columns, fp32) and exp(S) (f16) persist in registers.
// ---------------------------------------------------------------------------
__global__ __launch_bounds__(64, 1) void realnvp_fused_kernel(
    const float* __restrict__ z,
    const float* __restrict__ bs1, const float* __restrict__ bs2,
    const float* __restrict__ bt1, const float* __restrict__ bt2,
    const _Float16* __restrict__ wpk,
    float* __restrict__ out)
{
    __shared__ __align__(32) unsigned int s_e[2048];  //  8 KB: E, A-operand layout
    __shared__ __align__(32) unsigned int s_h[4096];  // 16 KB: H (lo) / G (hi) tiles
    __shared__ float s_lj[16];

    const int tid  = threadIdx.x;
    const int lane = tid & 31;
    const int wp   = (tid >> 5) & 1;  // wave-in-pair: owns N-tiles nb = 2*st + wp
    const int nhat = lane & 15;
    const int half = lane >> 4;
    const int m    = lane & 15;
    const int rowbase = blockIdx.x * 16;

    if (tid < 16) s_lj[tid] = 0.0f;

    const u32x8* s_e8 = (const u32x8*)s_e;
    u32x8*       s_e8w = (u32x8*)s_e;
    const u32x8* s_h8 = (const u32x8*)s_h;
    _Float16*    s_hh = (_Float16*)s_h;

    // per-lane base pointer into the packed-weight arena (tile offsets become
    // compile-time immediates on the loads)
    const char* wlane = (const char*)wpk + lane * 32;

    // ---- E (even z columns) -> f16 A-operand tiles, parked in LDS ----
    {
        const float* zr = z + (size_t)(rowbase + m) * DDIM;
#pragma unroll
        for (int kb = 0; kb < 8; ++kb) {
            v16h ea;
#pragma unroll
            for (int p = 0; p < 8; ++p) {
                int k = ka_base(p, half) + 32 * kb;      // even
                float4 f = *(const float4*)(zr + 2 * k);  // z[2k..2k+3]
                ea[2 * p]     = (_Float16)f.x;            // e[k]   = z[2k]
                ea[2 * p + 1] = (_Float16)f.z;            // e[k+1] = z[2k+2]
            }
            s_e8w[kb * 32 + lane] = __builtin_bit_cast(u32x8, ea);
        }
    }

    // ---- O (odd z columns), fp32 D-format; wave owns tiles nb = 2*st + wp ----
    v8f Ov[8];
#pragma unroll
    for (int st = 0; st < 8; ++st) {
        int nglob = (st * 2 + wp) * 16 + nhat;
#pragma unroll
        for (int r = 0; r < 8; ++r)
            Ov[st][r] = z[(size_t)(rowbase + r + 8 * half) * DDIM + 2 * nglob + 1];
    }

    float ljp[8];
#pragma unroll
    for (int r = 0; r < 8; ++r) ljp[r] = 0.0f;

    v8h es8[8];  // exp(s) per owned N-tile, packed f16

#pragma unroll 1
    for (int i = 0; i < NLAYER; ++i) {
        const char* pW1 = wlane + (size_t)(i * 4 + 0) * 131072;
        const char* pW2 = wlane + (size_t)(i * 4 + 1) * 131072;
        const char* pV1 = wlane + (size_t)(i * 4 + 2) * 131072;
        const char* pV2 = wlane + (size_t)(i * 4 + 3) * 131072;
        const float* B1 = bs1 + i * NHID;
        const float* B2 = bs2 + i * NHALF;
        const float* C1 = bt1 + i * NHID;
        const float* C2 = bt2 + i * NHALF;

        // ---- H = relu(E@Ws1 + b1) -> s_h[lo];  G = relu(E@Wt1 + c1) -> s_h[hi]
#pragma unroll
        for (int phg = 0; phg < 2; ++phg) {
            const char*  pW   = phg ? pV1 : pW1;
            const float* bia  = phg ? C1 : B1;
            const int    hbase = phg ? 4096 : 0;  // halves
#pragma unroll 2
            for (int st = 0; st < 8; ++st) {
                int nb = st * 2 + wp;
                v8f acc = {};
#pragma unroll
                for (int kb = 0; kb < 8; ++kb) {
                    v16h a = load_lds_a(s_e8, kb * 32 + lane);
                    v16h b = load_bw(pW, (nb * 8 + kb) * 1024);
                    acc = __builtin_amdgcn_wmma_f32_16x16x32_f16(
                        false, a, false, b, (short)0, acc, false, false);
                }
                int   kg   = nb * 16 + nhat;   // k index of the next GEMM
                float bias = bia[kg];
                // A-operand address decomposition for kg (constant per st)
                int kb2  = kg >> 5, kl = kg & 31;
                int hA   = (kl >> 3) & 1;
                int p    = ((kl & 7) >> 1) + ((kl >> 4) << 2);
                int lohi = kl & 1;
                int abase = hbase + (((kb2 * 32 + 16 * hA) * 8 + p) << 1) + lohi;
#pragma unroll
                for (int r = 0; r < 8; ++r) {
                    float v = fmaxf(acc[r] + bias, 0.0f);
                    int mm = r + 8 * half;
                    s_hh[abase + mm * 16] = (_Float16)v;
                }
            }
        }
        __syncthreads();

        // ---- S = tanh(H@Ws2 + b2): logjac += S, es = exp(S) ----
#pragma unroll
        for (int st = 0; st < 8; ++st) {
            int nb = st * 2 + wp;
            v8f acc = {};
#pragma unroll
            for (int kb = 0; kb < 8; ++kb) {
                v16h a = load_lds_a(s_h8, kb * 32 + lane);
                v16h b = load_bw(pW2, (nb * 8 + kb) * 1024);
                acc = __builtin_amdgcn_wmma_f32_16x16x32_f16(
                    false, a, false, b, (short)0, acc, false, false);
            }
            float bias = B2[nb * 16 + nhat];
#pragma unroll
            for (int r = 0; r < 8; ++r) {
                float s = tanh_fast(acc[r] + bias);
                ljp[r] += s;
                es8[st][r] = (_Float16)__expf(s);
            }
        }

        // ---- T = G@Wt2 + c2;  O = exp(S)*O + T ----
#pragma unroll
        for (int st = 0; st < 8; ++st) {
            int nb = st * 2 + wp;
            v8f acc = {};
#pragma unroll
            for (int kb = 0; kb < 8; ++kb) {
                v16h a = load_lds_a(s_h8, 256 + kb * 32 + lane);
                v16h b = load_bw(pV2, (nb * 8 + kb) * 1024);
                acc = __builtin_amdgcn_wmma_f32_16x16x32_f16(
                    false, a, false, b, (short)0, acc, false, false);
            }
            float bias = C2[nb * 16 + nhat];
#pragma unroll
            for (int r = 0; r < 8; ++r) {
                float t = acc[r] + bias;
                Ov[st][r] = (float)es8[st][r] * Ov[st][r] + t;
            }
        }
        __syncthreads();  // next layer overwrites s_h
    }

    // ---- Output: transformed odd half ----
#pragma unroll
    for (int st = 0; st < 8; ++st) {
        int nglob = (st * 2 + wp) * 16 + nhat;
#pragma unroll
        for (int r = 0; r < 8; ++r)
            out[(size_t)(rowbase + r + 8 * half) * DDIM + NHALF + nglob] = Ov[st][r];
    }

    // ---- Output: untransformed even half, bit-exact copy from z ----
    if (wp == 0) {
#pragma unroll 1
        for (int r = 0; r < 16; ++r) {
            const float* zr = z   + (size_t)(rowbase + r) * DDIM;
            float* orow     = out + (size_t)(rowbase + r) * DDIM;
#pragma unroll
            for (int c = 0; c < 8; ++c) {
                int col = c * 32 + lane;
                orow[col] = zr[2 * col];
            }
        }
    }

    // ---- logjac: reduce over the 16 column-lanes of each half, then waves ----
#pragma unroll
    for (int r = 0; r < 8; ++r) {
        float v = ljp[r];
        v += __shfl_xor(v, 1, 32);
        v += __shfl_xor(v, 2, 32);
        v += __shfl_xor(v, 4, 32);
        v += __shfl_xor(v, 8, 32);
        if (nhat == 0) atomicAdd(&s_lj[r + 8 * half], v);
    }
    __syncthreads();
    if (tid < 16)
        out[(size_t)NBATCH * DDIM + blockIdx.x * 16 + tid] = s_lj[tid];
}

extern "C" void kernel_launch(void* const* d_in, const int* in_sizes, int n_in,
                              void* d_out, int out_size, void* d_ws, size_t ws_size,
                              hipStream_t stream) {
    (void)in_sizes; (void)n_in; (void)ws_size; (void)out_size;
    const float* z   = (const float*)d_in[0];
    const float* Ws1 = (const float*)d_in[1];
    const float* bs1 = (const float*)d_in[2];
    const float* Ws2 = (const float*)d_in[3];
    const float* bs2 = (const float*)d_in[4];
    const float* Wt1 = (const float*)d_in[5];
    const float* bt1 = (const float*)d_in[6];
    const float* Wt2 = (const float*)d_in[7];
    const float* bt2 = (const float*)d_in[8];
    // d_in[9]/d_in[10] (yidx/xidx) encode the fixed even/odd masks, folded in.
    float* out = (float*)d_out;

    _Float16* wpk = (_Float16*)d_ws;  // needs 4 MiB of workspace

    // Pack all 32 weight matrices into WMMA B-operand layout (f16, L2-resident).
    pack_weights_kernel<<<dim3(2048), dim3(256), 0, stream>>>(Ws1, Ws2, Wt1, Wt2, wpk);

    // Fused flow: one wave pair per 16 rows.
    realnvp_fused_kernel<<<dim3(NBATCH / 16), dim3(64), 0, stream>>>(
        z, bs1, bs2, bt1, bt2, wpk, out);
}
